// CausalSelfAttention_2903397892100
// MI455X (gfx1250) — matrix operand order repaired and checked by reading, not compiled, not run on hardware
//
#include <hip/hip_runtime.h>
#include <hip/hip_bf16.h>
#include <stdint.h>

typedef __attribute__((ext_vector_type(16))) __bf16 v16bf;
typedef __attribute__((ext_vector_type(8)))  float  v8f;
typedef __attribute__((ext_vector_type(4)))  unsigned int v4u;
typedef __attribute__((ext_vector_type(8)))  int v8i;
typedef __attribute__((ext_vector_type(4)))  int v4i;

namespace {
constexpr int BB = 2, TT = 2048, CC = 1024, HH = 16, HD = 64;
constexpr int M_ROWS = BB * TT;   // 4096
constexpr int N_QKV  = 3 * CC;    // 3072
}

union AFrag { v16bf v; unsigned int u[8]; };

// A-fragment (16-bit, 16x32) K base for register pair r, lane-half h (contiguous k = 16h + 2r, matching B fills)
__device__ __forceinline__ int kbaseA(int r, int h) {
  return 16*h + 2*r;
}

__device__ __forceinline__ unsigned short f2bf(float f) {
  unsigned int u = __float_as_uint(f);
  u += 0x7fffu + ((u >> 16) & 1u);   // round-to-nearest-even
  return (unsigned short)(u >> 16);
}

__device__ __forceinline__ v8f wmma_bf16(const AFrag& a, const AFrag& b, v8f c) {
  return __builtin_amdgcn_wmma_f32_16x16x32_bf16(false, a.v, false, b.v,
                                                 (short)0, c, false, false);
}

#if __has_builtin(__builtin_amdgcn_tensor_load_to_lds)
#define HAS_TDM 1
#else
#define HAS_TDM 0
#endif

__device__ __forceinline__ void wait_tensorcnt0() {
#if __has_builtin(__builtin_amdgcn_s_wait_tensorcnt)
  __builtin_amdgcn_s_wait_tensorcnt(0);
#else
  asm volatile("s_wait_tensorcnt 0" ::: "memory");
#endif
}

// 2-D TDM tile load: tensor is [tensor_d1 rows][tensor_d0 cols] of 2-byte
// elements, row-major; copies a [rows][cols] tile starting at gbase into LDS
// contiguously (D# per CDNA5 ISA sec. 8.3/8.4). This toolchain exposes the
// 6-arg builtin: (v4u g0, v8i g1, v4i g2, v4i g3, v8i extra, i32 cpol).
__device__ __forceinline__ void tdm_load_tile_2d(const unsigned short* gbase,
                                                 unsigned int lds_off,
                                                 int rows, int cols,
                                                 int tensor_d0, int tensor_d1) {
#if HAS_TDM
  unsigned long long ga = (unsigned long long)(uintptr_t)gbase;
  union { unsigned int d[4]; v4u v; } g0;
  g0.d[0] = 1u;                                            // count=1 (valid)
  g0.d[1] = lds_off;                                       // lds_addr
  g0.d[2] = (unsigned int)(ga & 0xffffffffull);            // global_addr lo
  g0.d[3] = (unsigned int)((ga >> 32) & 0x1ffffffull)      // global_addr hi
          | (2u << 30);                                    // type = 2 ("image")
  union { unsigned long long q[4]; v8i v; } g1;
  unsigned long long d0 = (unsigned)tensor_d0, d1 = (unsigned)tensor_d1;
  g1.q[0] = (1ull << 16)                                   // data_size = 2 B
          | ((d0 & 0xffffull) << 48);                      // tensor_dim0 [15:0]
  g1.q[1] = (d0 >> 16)                                     // tensor_dim0 [31:16]
          | (d1 << 16)                                     // tensor_dim1
          | ((unsigned long long)(unsigned)cols << 48);    // tile_dim0
  g1.q[2] = (unsigned long long)(unsigned)rows             // tile_dim1
          | ((d0 & 0xffffffffull) << 32);                  // dim0_stride [31:0]
  g1.q[3] = 0ull;                                          // stride hi / stride1
  v4i z4 = {0, 0, 0, 0};
  v8i z8 = {0, 0, 0, 0, 0, 0, 0, 0};
  __builtin_amdgcn_tensor_load_to_lds(g0.v, g1.v, z4, z4, z8, 0);
#else
  (void)gbase; (void)lds_off; (void)rows; (void)cols; (void)tensor_d0; (void)tensor_d1;
#endif
}

// ---------------------------------------------------------------- converts
__global__ void f32_to_bf16_kernel(const float* __restrict__ src,
                                   unsigned short* __restrict__ dst, int n) {
  int i = blockIdx.x * blockDim.x + threadIdx.x;
  if (i < n) dst[i] = f2bf(src[i]);
}

// W [K][N] fp32 -> W^T [N][K] bf16 (tile transpose through LDS, coalesced)
__global__ __launch_bounds__(256) void transpose_f32_to_bf16(
    const float* __restrict__ src, unsigned short* __restrict__ dst,
    int K, int N) {
  __shared__ unsigned short tile[32][33];
  const int kt = blockIdx.y * 32, nt = blockIdx.x * 32;
  const int tx = threadIdx.x & 31, ty = threadIdx.x >> 5;   // ty = 0..7
#pragma unroll
  for (int i = ty; i < 32; i += 8)
    tile[i][tx] = f2bf(src[(size_t)(kt + i) * N + nt + tx]);
  __syncthreads();
#pragma unroll
  for (int i = ty; i < 32; i += 8)
    dst[(size_t)(nt + i) * K + kt + tx] = tile[tx][i];
}

// ------------------------------------------------------------- GEMM core
// Wave computes 32(M) x 64(N); block = 4 waves = 128x64 tile.
// WT is the pre-transposed weight [Ndim][Kdim] bf16; a 64(N) x 32(K) tile is
// TDM-copied into LDS each K step (layout [n][k], K-pairs contiguous => the
// B-fragment reads are plain ds_load_b128).
__device__ __forceinline__ void gemm_mainloop(const unsigned short* __restrict__ Abf,
                                              const unsigned short* __restrict__ WT,
                                              int Kdim, int Ndim,
                                              int mbase, int nbase,
                                              int wave, int lane,
                                              unsigned short* wtile, v8f acc[2][4]) {
  const int hh = lane >> 4, ln = lane & 15;
  const int arow0 = mbase + wave * 32 + ln;
  const unsigned int lds_off = (unsigned int)(uintptr_t)wtile;
  for (int kc = 0; kc < Kdim; kc += 32) {
    __syncthreads();
#if HAS_TDM
    if (wave == 0) {
      tdm_load_tile_2d(WT + (size_t)nbase * Kdim + kc, lds_off,
                       /*rows=*/64, /*cols=*/32, Kdim, Ndim);
      wait_tensorcnt0();
    }
#else
    for (int e = threadIdx.x; e < 64 * 32; e += blockDim.x) {
      int n = e >> 5, kk = e & 31;
      wtile[n * 32 + kk] = WT[(size_t)(nbase + n) * Kdim + kc + kk];
    }
#endif
    __syncthreads();
    AFrag a0, a1;
#pragma unroll
    for (int r = 0; r < 8; ++r) {
      a0.u[r] = *(const unsigned int*)(Abf + (size_t)arow0 * Kdim + kc + kbaseA(r, hh));
      a1.u[r] = *(const unsigned int*)(Abf + (size_t)(arow0 + 16) * Kdim + kc + kbaseA(r, hh));
    }
#pragma unroll
    for (int j = 0; j < 4; ++j) {
      AFrag b;
#pragma unroll
      for (int r = 0; r < 8; ++r)
        b.u[r] = *(const unsigned int*)(wtile + (16 * j + ln) * 32 + 16 * hh + 2 * r);
      acc[0][j] = wmma_bf16(a0, b, acc[0][j]);
      acc[1][j] = wmma_bf16(a1, b, acc[1][j]);
    }
  }
}

// --------------------------------------------------------------- QKV GEMM
// q,k stored [B,H,T,HD]; v stored transposed [B,H,HD,T] for the PV WMMA.
__global__ __launch_bounds__(128) void qkv_gemm_kernel(
    const unsigned short* __restrict__ xb, const unsigned short* __restrict__ waT,
    const float* __restrict__ bias,
    unsigned short* __restrict__ qd, unsigned short* __restrict__ kd,
    unsigned short* __restrict__ vd) {
  __shared__ unsigned short wtile[64 * 32];
  const int nbase = blockIdx.x * 64, mbase = blockIdx.y * 128;
  const int wave = threadIdx.x >> 5, lane = threadIdx.x & 31;
  const int hh = lane >> 4, ln = lane & 15;
  v8f acc[2][4] = {};
  gemm_mainloop(xb, waT, CC, N_QKV, mbase, nbase, wave, lane, wtile, acc);
#pragma unroll
  for (int j = 0; j < 4; ++j) {
    const int n = nbase + 16 * j + ln;
    const float bv = bias[n];
    const int mat = n / CC, rem = n % CC;
    const int head = rem / HD, d = rem % HD;
#pragma unroll
    for (int s = 0; s < 2; ++s) {
#pragma unroll
      for (int r = 0; r < 8; ++r) {
        const int m = mbase + wave * 32 + s * 16 + r + 8 * hh;  // C-frag row
        const int b = m / TT, t = m % TT;
        const unsigned short val = f2bf(acc[s][j][r] + bv);
        if (mat == 2)
          vd[(((size_t)b * HH + head) * HD + d) * TT + t] = val;       // V^T
        else if (mat == 1)
          kd[(((size_t)b * HH + head) * TT + t) * HD + d] = val;
        else
          qd[(((size_t)b * HH + head) * TT + t) * HD + d] = val;
      }
    }
  }
}

// ---------------------------------------------------------- flash attention
// One wave = 16 query rows; block = 4 waves = 64 rows per (b,head).
__global__ __launch_bounds__(128) void flash_attn_kernel(
    const unsigned short* __restrict__ qg, const unsigned short* __restrict__ kg,
    const unsigned short* __restrict__ vg, unsigned short* __restrict__ y) {
  __shared__ unsigned short ptile[4][16 * 32];   // per-wave P scratch (bf16)
  const int bh = blockIdx.y;
  const int b = bh / HH, head = bh % HH;
  const int wave = threadIdx.x >> 5, lane = threadIdx.x & 31;
  const int hh = lane >> 4, ln = lane & 15;
  const int qrow0 = blockIdx.x * 64 + wave * 16;
  const unsigned short* qp  = qg + (size_t)bh * TT * HD;
  const unsigned short* kp  = kg + (size_t)bh * TT * HD;
  const unsigned short* vpT = vg + (size_t)bh * HD * TT;   // V^T [HD][T]
  unsigned short* pt = ptile[wave];

  // Q as two A-fragments (dims 0..31, 32..63)
  AFrag aQ[2];
  {
    const int row = qrow0 + ln;
#pragma unroll
    for (int c = 0; c < 2; ++c)
#pragma unroll
      for (int r = 0; r < 8; ++r)
        aQ[c].u[r] = *(const unsigned int*)(qp + (size_t)row * HD + 32 * c + kbaseA(r, hh));
  }

  v8f o[4] = {};
  float mrow[8], lrow[8];
#pragma unroll
  for (int r = 0; r < 8; ++r) { mrow[r] = -__builtin_inff(); lrow[r] = 0.f; }
  const float scale = 0.125f;          // 1/sqrt(64)
  const int qmax = qrow0 + 15;

  for (int kt = 0; kt <= qmax; kt += 32) {
    // S = Q * K^T for 32 keys -> two 16x16 fp32 tiles
    v8f s[2] = {};
#pragma unroll
    for (int jc = 0; jc < 2; ++jc) {
      const int key = kt + 16 * jc + ln;
#pragma unroll
      for (int c = 0; c < 2; ++c) {
        AFrag bk;   // B-frag: N=key, K=head-dim (dim pairs contiguous)
#pragma unroll
        for (int r = 0; r < 8; ++r)
          bk.u[r] = *(const unsigned int*)(kp + (size_t)key * HD + 32 * c + 16 * hh + 2 * r);
        s[jc] = wmma_bf16(aQ[c], bk, s[jc]);
      }
    }

    // online softmax: row stats live per-lane for the 8 rows (r + 8*hh)
#pragma unroll
    for (int r = 0; r < 8; ++r) {
      const int qrow = qrow0 + r + 8 * hh;
      float a0 = s[0][r] * scale;
      float a1 = s[1][r] * scale;
      if (kt + ln      > qrow) a0 = -__builtin_inff();   // causal mask
      if (kt + 16 + ln > qrow) a1 = -__builtin_inff();
      float mx = fmaxf(a0, a1);
#pragma unroll
      for (int msk = 8; msk >= 1; msk >>= 1) mx = fmaxf(mx, __shfl_xor(mx, msk, 16));
      const float mnew = fmaxf(mrow[r], mx);
      const float sc   = __expf(mrow[r] - mnew);
      mrow[r] = mnew;
      const float p0 = __expf(a0 - mnew);
      const float p1 = __expf(a1 - mnew);
      float rs = p0 + p1;
#pragma unroll
      for (int msk = 8; msk >= 1; msk >>= 1) rs += __shfl_xor(rs, msk, 16);
      lrow[r] = lrow[r] * sc + rs;
      pt[(r + 8 * hh) * 32 + ln]      = f2bf(p0);   // C-frag -> LDS
      pt[(r + 8 * hh) * 32 + 16 + ln] = f2bf(p1);
#pragma unroll
      for (int j = 0; j < 4; ++j) o[j][r] *= sc;     // rescale O
    }
    asm volatile("s_wait_dscnt 0" ::: "memory");     // wave-local LDS RAW fence

    // reload P as A-fragment (16 x 32 keys)
    AFrag pf;
#pragma unroll
    for (int r = 0; r < 8; ++r)
      pf.u[r] = *(const unsigned int*)(pt + ln * 32 + kbaseA(r, hh));

    // O += P * V   (V^T: key pairs contiguous -> b128-mergeable loads)
#pragma unroll
    for (int j = 0; j < 4; ++j) {
      AFrag bv;
      const unsigned short* vrow = vpT + (size_t)(16 * j + ln) * TT + kt + 16 * hh;
#pragma unroll
      for (int r = 0; r < 8; ++r)
        bv.u[r] = *(const unsigned int*)(vrow + 2 * r);
      o[j] = wmma_bf16(pf, bv, o[j]);
    }
  }

  // normalize + store y[B,T,C] as bf16
#pragma unroll
  for (int r = 0; r < 8; ++r) {
    const float inv = 1.0f / lrow[r];
    const int t = qrow0 + r + 8 * hh;
#pragma unroll
    for (int j = 0; j < 4; ++j) {
      const int cidx = head * HD + 16 * j + ln;
      y[((size_t)b * TT + t) * CC + cidx] = f2bf(o[j][r] * inv);
    }
  }
}

// ------------------------------------------------------------ output proj
__global__ __launch_bounds__(128) void proj_gemm_kernel(
    const unsigned short* __restrict__ yb, const unsigned short* __restrict__ wpT,
    const float* __restrict__ bias, float* __restrict__ out) {
  __shared__ unsigned short wtile[64 * 32];
  const int nbase = blockIdx.x * 64, mbase = blockIdx.y * 128;
  const int wave = threadIdx.x >> 5, lane = threadIdx.x & 31;
  const int hh = lane >> 4, ln = lane & 15;
  v8f acc[2][4] = {};
  gemm_mainloop(yb, wpT, CC, CC, mbase, nbase, wave, lane, wtile, acc);
#pragma unroll
  for (int j = 0; j < 4; ++j) {
    const int n = nbase + 16 * j + ln;
    const float bv = bias[n];
#pragma unroll
    for (int s = 0; s < 2; ++s)
#pragma unroll
      for (int r = 0; r < 8; ++r) {
        const int m = mbase + wave * 32 + s * 16 + r + 8 * hh;
        out[(size_t)m * CC + n] = acc[s][j][r] + bv;
      }
  }
}

// ------------------------------------------------------------------ launch
extern "C" void kernel_launch(void* const* d_in, const int* in_sizes, int n_in,
                              void* d_out, int out_size, void* d_ws, size_t ws_size,
                              hipStream_t stream) {
  (void)in_sizes; (void)n_in; (void)out_size; (void)ws_size;
  const float* x      = (const float*)d_in[0];
  const float* w_attn = (const float*)d_in[1];
  const float* b_attn = (const float*)d_in[2];
  const float* w_proj = (const float*)d_in[3];
  const float* b_proj = (const float*)d_in[4];
  float* out = (float*)d_out;

  char* ws = (char*)d_ws;
  size_t off = 0;
  auto alloc = [&](size_t bytes) {
    char* p = ws + off;
    off = (off + bytes + 255) & ~(size_t)255;
    return (unsigned short*)p;
  };
  unsigned short* xb  = alloc((size_t)M_ROWS * CC * 2);   // x bf16 [M][K]
  unsigned short* waT = alloc((size_t)CC * N_QKV * 2);    // w_attn^T bf16 [3C][C]
  unsigned short* wpT = alloc((size_t)CC * CC * 2);       // w_proj^T bf16 [C][C]
  unsigned short* qb  = alloc((size_t)M_ROWS * CC * 2);   // q [B,H,T,HD]
  unsigned short* kb  = alloc((size_t)M_ROWS * CC * 2);   // k [B,H,T,HD]
  unsigned short* vb  = alloc((size_t)M_ROWS * CC * 2);   // v^T [B,H,HD,T]
  unsigned short* yb  = alloc((size_t)M_ROWS * CC * 2);   // attn out [B,T,C]

  const int n_x = M_ROWS * CC;
  f32_to_bf16_kernel<<<(n_x + 255) / 256, 256, 0, stream>>>(x, xb, n_x);
  transpose_f32_to_bf16<<<dim3(N_QKV / 32, CC / 32), 256, 0, stream>>>(
      w_attn, waT, CC, N_QKV);
  transpose_f32_to_bf16<<<dim3(CC / 32, CC / 32), 256, 0, stream>>>(
      w_proj, wpT, CC, CC);

  qkv_gemm_kernel<<<dim3(N_QKV / 64, M_ROWS / 128), 128, 0, stream>>>(
      xb, waT, b_attn, qb, kb, vb);

  flash_attn_kernel<<<dim3(TT / 64, BB * HH), 128, 0, stream>>>(qb, kb, vb, yb);

  proj_gemm_kernel<<<dim3(CC / 64, M_ROWS / 128), 128, 0, stream>>>(
      yb, wpT, b_proj, out);
}
